// UCR_53833120088480
// MI455X (gfx1250) — matrix operand order repaired and checked by reading, not compiled, not run on hardware
//
#include <hip/hip_runtime.h>
#include <hip/hip_bf16.h>

typedef _Float16 v4h  __attribute__((ext_vector_type(4)));
typedef _Float16 v8h  __attribute__((ext_vector_type(8)));
typedef _Float16 v16h __attribute__((ext_vector_type(16)));
typedef float    v8f  __attribute__((ext_vector_type(8)));

constexpr int   kN        = 8192;
constexpr int   kD        = 1024;
constexpr float kKeep     = 0.9f;
constexpr float kInvKeep  = 1.0f / 0.9f;
constexpr float kEps      = 1e-8f;
// logits = sim / 0.1 ; we work in base-2 domain: y = sim * (1/TEMP) * log2(e)
constexpr float kScale2   = 10.0f * 1.44269504088896340736f;
constexpr float kLn2      = 0.69314718055994530942f;
constexpr float kNegBig   = -3.0e38f;

constexpr int kMBlk   = 128;    // rows per block
constexpr int kNBlk   = 64;     // cols per iteration
constexpr int kKC     = 128;    // K chunk staged in LDS
constexpr int kLdsStr = kKC + 8;  // 136 halves -> 272B row stride (68 banks): conflict-friendly
constexpr int kColSplit = 4;    // column range split across blocks for occupancy
constexpr int kNPart  = kColSplit * 2;  // 2 col-half waves per split produce partials

// ---------------------------------------------------------------------------
// Kernel 1: dropout + row norms + normalize, emit fp16 a (from inputs) and
// b (from enhanced), row-major. b rows == WMMA B-matrix columns (K-contiguous).
// ---------------------------------------------------------------------------
__global__ __launch_bounds__(256) void prep_kernel(const float* __restrict__ inp,
                                                   const float* __restrict__ noi,
                                                   _Float16* __restrict__ a16,
                                                   _Float16* __restrict__ b16) {
  const int row = blockIdx.x;
  const int t   = threadIdx.x;

  const float4 x = ((const float4*)(inp + (size_t)row * kD))[t];
  const float4 n = ((const float4*)(noi + (size_t)row * kD))[t];
  float4 e;
  e.x = x.x * (n.x < kKeep ? kInvKeep : 0.0f);
  e.y = x.y * (n.y < kKeep ? kInvKeep : 0.0f);
  e.z = x.z * (n.z < kKeep ? kInvKeep : 0.0f);
  e.w = x.w * (n.w < kKeep ? kInvKeep : 0.0f);

  float sa = x.x * x.x + x.y * x.y + x.z * x.z + x.w * x.w;
  float sb = e.x * e.x + e.y * e.y + e.z * e.z + e.w * e.w;
  #pragma unroll
  for (int msk = 1; msk <= 16; msk <<= 1) {
    sa += __shfl_xor(sa, msk, 32);
    sb += __shfl_xor(sb, msk, 32);
  }

  __shared__ float red[2][8];
  __shared__ float scl[2];
  const int wave = t >> 5, lane = t & 31;
  if (lane == 0) { red[0][wave] = sa; red[1][wave] = sb; }
  __syncthreads();
  if (t == 0) {
    float ta = 0.0f, tb = 0.0f;
    #pragma unroll
    for (int w = 0; w < 8; ++w) { ta += red[0][w]; tb += red[1][w]; }
    scl[0] = 1.0f / fmaxf(sqrtf(ta), kEps);
    scl[1] = 1.0f / fmaxf(sqrtf(tb), kEps);
  }
  __syncthreads();
  const float ia = scl[0], ib = scl[1];

  v4h av = { (_Float16)(x.x * ia), (_Float16)(x.y * ia),
             (_Float16)(x.z * ia), (_Float16)(x.w * ia) };
  v4h bv = { (_Float16)(e.x * ib), (_Float16)(e.y * ib),
             (_Float16)(e.z * ib), (_Float16)(e.w * ib) };
  *(v4h*)(a16 + (size_t)row * kD + t * 4) = av;
  *(v4h*)(b16 + (size_t)row * kD + t * 4) = bv;
}

// ---------------------------------------------------------------------------
// Fragment loads from LDS following the documented 16-bit WMMA VGPR striping.
// A 16x32 f16: lanes 0-15 hold M=lane, halves {K0..K0+7, K0+16..K0+23};
//              lanes 16-31 hold M=lane-16, halves {K0+8..15, K0+24..31}.
// B 32x16 f16: lane%16 = N, same K striping as A (lo half K0+8h..+7, K0+16+8h..+7).
// ---------------------------------------------------------------------------
__device__ inline v16h load_frag_a(const _Float16* p) {
  v8h lo = *(const v8h*)p;
  v8h hi = *(const v8h*)(p + 16);
  return __builtin_shufflevector(lo, hi, 0,1,2,3,4,5,6,7,8,9,10,11,12,13,14,15);
}
__device__ inline v16h load_frag_b(const _Float16* p) {
  v8h lo = *(const v8h*)p;
  v8h hi = *(const v8h*)(p + 16);
  return __builtin_shufflevector(lo, hi, 0,1,2,3,4,5,6,7,8,9,10,11,12,13,14,15);
}

// ---------------------------------------------------------------------------
// Kernel 2: fused a @ b^T / TEMP with online base-2 logsumexp.
// grid = (kColSplit, kN/kMBlk); block = 256 (8 waves = 4 row-tiles x 2 col-halves).
// Each wave owns a 32x32 register tile: 4 f32 accumulators fed by 2 A + 2 B
// fragments per K-step -> 2 ds_load_b128 per WMMA (vs 3 in the 16x32 scheme).
// ---------------------------------------------------------------------------
__global__ __launch_bounds__(256) void logits_lse_kernel(
    const _Float16* __restrict__ a16, const _Float16* __restrict__ b16,
    float* __restrict__ mPart, float* __restrict__ sPart,
    float* __restrict__ ydiag) {
  __shared__ _Float16 As[kMBlk][kLdsStr];
  __shared__ _Float16 Bs[kNBlk][kLdsStr];

  const int tid  = threadIdx.x;
  const int lane = tid & 31;
  const int wave = tid >> 5;
  const int wm   = wave & 3;   // row tile pair 0..3 (32 rows each)
  const int wn   = wave >> 2;  // col half 0..1 (32 cols each)
  const int l15  = lane & 15;
  const int hi   = lane >> 4;
  const int hi8  = hi * 8;

  const int rowBase      = blockIdx.y * kMBlk;
  const int colSplitBase = blockIdx.x * (kN / kColSplit);
  const int waveRowT0    = rowBase + wm * 32;        // first 16-row tile
  const int waveRowT1    = waveRowT0 + 16;           // second 16-row tile

  float m[2][8], s[2][8];
  #pragma unroll
  for (int g = 0; g < 2; ++g)
    #pragma unroll
    for (int r = 0; r < 8; ++r) { m[g][r] = kNegBig; s[g][r] = 0.0f; }

  for (int cb = 0; cb < (kN / kColSplit) / kNBlk; ++cb) {
    const int colBase = colSplitBase + cb * kNBlk;
    v8f c00 = {}, c01 = {}, c10 = {}, c11 = {};

    for (int kc = 0; kc < kD; kc += kKC) {
      __syncthreads();  // previous chunk's compute must finish before overwrite
      // cooperative stage: A(128x128) + B(64x128) halves, 16B units
      #pragma unroll
      for (int u = tid; u < kMBlk * (kKC / 8); u += 256) {
        const int r  = u >> 4;            // 0..127
        const int kk = (u & 15) << 3;     // 0..120 step 8
        *(v8h*)&As[r][kk] = *(const v8h*)(a16 + (size_t)(rowBase + r) * kD + kc + kk);
      }
      #pragma unroll
      for (int u = tid; u < kNBlk * (kKC / 8); u += 256) {
        const int r  = u >> 4;            // 0..63
        const int kk = (u & 15) << 3;
        *(v8h*)&Bs[r][kk] = *(const v8h*)(b16 + (size_t)(colBase + r) * kD + kc + kk);
      }
      __syncthreads();

      #pragma unroll
      for (int k0 = 0; k0 < kKC; k0 += 32) {
        v16h a0 = load_frag_a(&As[wm * 32 + l15][k0 + hi8]);
        v16h a1 = load_frag_a(&As[wm * 32 + 16 + l15][k0 + hi8]);
        v16h b0 = load_frag_b(&Bs[wn * 32 + l15][k0 + hi8]);
        v16h b1 = load_frag_b(&Bs[wn * 32 + 16 + l15][k0 + hi8]);
        c00 = __builtin_amdgcn_wmma_f32_16x16x32_f16(false, a0, false, b0,
                                                     (short)0, c00, false, false);
        c01 = __builtin_amdgcn_wmma_f32_16x16x32_f16(false, a0, false, b1,
                                                     (short)0, c01, false, false);
        c10 = __builtin_amdgcn_wmma_f32_16x16x32_f16(false, a1, false, b0,
                                                     (short)0, c10, false, false);
        c11 = __builtin_amdgcn_wmma_f32_16x16x32_f16(false, a1, false, b1,
                                                     (short)0, c11, false, false);
      }
    }

    // ---- online logsumexp epilogue (base-2 domain) + diagonal capture ----
    const int ct0 = colBase + wn * 32;   // first col tile; second is ct0+16
    #pragma unroll
    for (int g = 0; g < 2; ++g) {
      const int  rowT = (g == 0) ? waveRowT0 : waveRowT1;
      const bool d0   = (ct0 == rowT);
      const bool d1   = (ct0 + 16 == rowT);
      const v8f& cA   = (g == 0) ? c00 : c10;
      const v8f& cB   = (g == 0) ? c01 : c11;
      #pragma unroll
      for (int r = 0; r < 8; ++r) {
        const float y0 = cA[r] * kScale2;
        const float y1 = cB[r] * kScale2;
        if (d0 && l15 == r + hi8) ydiag[rowT + r + hi8] = y0;
        if (d1 && l15 == r + hi8) ydiag[rowT + r + hi8] = y1;
        const float nm = fmaxf(m[g][r], fmaxf(y0, y1));
        s[g][r] = s[g][r] * exp2f(m[g][r] - nm) + exp2f(y0 - nm) + exp2f(y1 - nm);
        m[g][r] = nm;
      }
    }
  }

  // cross-lane merge within the 16-lane half that shares rows; emit partials
  const int part = blockIdx.x * 2 + wn;
  #pragma unroll
  for (int g = 0; g < 2; ++g) {
    const int rowT = (g == 0) ? waveRowT0 : waveRowT1;
    #pragma unroll
    for (int r = 0; r < 8; ++r) {
      float mm = m[g][r], ss = s[g][r];
      #pragma unroll
      for (int msk = 1; msk <= 8; msk <<= 1) {
        const float mo = __shfl_xor(mm, msk, 32);
        const float so = __shfl_xor(ss, msk, 32);
        const float nm = fmaxf(mm, mo);
        ss = ss * exp2f(mm - nm) + so * exp2f(mo - nm);
        mm = nm;
      }
      if (l15 == 0) {
        const int row = rowT + r + hi8;
        mPart[(size_t)part * kN + row] = mm;
        sPart[(size_t)part * kN + row] = ss;
      }
    }
  }
}

// ---------------------------------------------------------------------------
// Kernel 3: merge per-split partials, subtract diagonal, mean -> scalar.
// ---------------------------------------------------------------------------
__global__ __launch_bounds__(256) void finalize_kernel(
    const float* __restrict__ mPart, const float* __restrict__ sPart,
    const float* __restrict__ ydiag, float* __restrict__ out) {
  float acc = 0.0f;
  for (int row = threadIdx.x; row < kN; row += 256) {
    float M = mPart[row], S = sPart[row];
    #pragma unroll
    for (int p = 1; p < kNPart; ++p) {
      const float m2 = mPart[(size_t)p * kN + row];
      const float s2 = sPart[(size_t)p * kN + row];
      const float nm = fmaxf(M, m2);
      S = S * exp2f(M - nm) + s2 * exp2f(m2 - nm);
      M = nm;
    }
    acc += (M + log2f(S)) - ydiag[row];   // (lse - diag) in base-2 units
  }
  #pragma unroll
  for (int msk = 1; msk <= 16; msk <<= 1) acc += __shfl_xor(acc, msk, 32);
  __shared__ float red[8];
  if ((threadIdx.x & 31) == 0) red[threadIdx.x >> 5] = acc;
  __syncthreads();
  if (threadIdx.x == 0) {
    float t = 0.0f;
    #pragma unroll
    for (int w = 0; w < 8; ++w) t += red[w];
    out[0] = t * (kLn2 / (float)kN);      // convert base-2 -> ln, mean
  }
}

// ---------------------------------------------------------------------------
extern "C" void kernel_launch(void* const* d_in, const int* in_sizes, int n_in,
                              void* d_out, int out_size, void* d_ws, size_t ws_size,
                              hipStream_t stream) {
  const float* inp = (const float*)d_in[0];
  const float* noi = (const float*)d_in[1];

  char* ws = (char*)d_ws;
  _Float16* a16 = (_Float16*)ws;
  _Float16* b16 = a16 + (size_t)kN * kD;
  float* mPart = (float*)(ws + 2 * (size_t)kN * kD * sizeof(_Float16));
  float* sPart = mPart + (size_t)kNPart * kN;
  float* ydiag = sPart + (size_t)kNPart * kN;
  float* out   = (float*)d_out;

  prep_kernel<<<kN, 256, 0, stream>>>(inp, noi, a16, b16);
  logits_lse_kernel<<<dim3(kColSplit, kN / kMBlk), 256, 0, stream>>>(
      a16, b16, mPart, sPart, ydiag);
  finalize_kernel<<<1, 256, 0, stream>>>(mPart, sPart, ydiag, out);
}
